// GATLayerEdgeAverage_82197084111207
// MI455X (gfx1250) — matrix operand order repaired and checked by reading, not compiled, not run on hardware
//
#include <hip/hip_runtime.h>

#define N_NODES 50000
#define N_EDGES 800000
#define DIM_I   64
#define DIM_O   64
#define KDIM    128                     // 2 * DIM_I
#define GAT_EPS 1e-6f

#define NUM_WAVES       8               // waves per block
#define GRID_BLKS       1250
#define TOT_WAVES       (GRID_BLKS * NUM_WAVES)      // 10000
#define NUM_TILES       (N_EDGES / 16)               // 50000
#define TILES_PER_WAVE  (NUM_TILES / TOT_WAVES)      // 5 (exact)

#define XH_BYTES        ((size_t)N_NODES * DIM_I * 2)   // 6.4 MB f16 copy of x
#define A_BYTES         ((size_t)N_EDGES * 4)           // 3.2 MB gate values

typedef __attribute__((ext_vector_type(16))) _Float16 v16h;
typedef __attribute__((ext_vector_type(8)))  _Float16 v8h;
typedef __attribute__((ext_vector_type(8)))  float    v8f;

// ---------------------------------------------------------------------------
// Kernel 1: zero accumulators (numerator in d_out, denominator in ws)
// ---------------------------------------------------------------------------
__global__ void gat_init(float* __restrict__ out, float* __restrict__ denom) {
    int i = blockIdx.x * blockDim.x + threadIdx.x;
    if (i < N_NODES * DIM_O) out[i] = 0.f;
    if (i < N_NODES)         denom[i] = 0.f;
}

// ---------------------------------------------------------------------------
// Kernel 2: x (f32) -> xh (f16), once; gather then feeds WMMA A-operand
// registers directly from global 16B chunks (halves L2 gather bytes).
// ---------------------------------------------------------------------------
__global__ void gat_x2h(const float* __restrict__ x, _Float16* __restrict__ xh) {
    int i = blockIdx.x * blockDim.x + threadIdx.x;
    if (i < N_NODES * DIM_I) xh[i] = (_Float16)x[i];
}

// ---------------------------------------------------------------------------
// Kernel 3: gate  a[e] = [xh[src]|xh[tgt]] . Ww + bw ; denom = segsum(a, tgt).
// 2 lanes per edge (even=src half, odd=tgt half), combine via shfl_xor(1).
// All reads hit the L2-resident xh; this also absorbs the denom atomics so
// the WMMA kernel stays lean.
// ---------------------------------------------------------------------------
__global__ __launch_bounds__(256)
void gat_gate(const _Float16* __restrict__ xh,
              const int*   __restrict__ src,
              const int*   __restrict__ tgt,
              const float* __restrict__ Ww,
              const float* __restrict__ bw,
              float* __restrict__ a_arr,
              float* __restrict__ denom) {
    __shared__ __attribute__((aligned(16))) _Float16 sWw[KDIM];
    if (threadIdx.x < KDIM) sWw[threadIdx.x] = (_Float16)Ww[threadIdx.x];
    __syncthreads();

    const int t    = blockIdx.x * 256 + threadIdx.x;  // 2*E threads total
    const int e    = t >> 1;
    const int half = t & 1;
    const int idx  = half ? tgt[e] : src[e];
    const _Float16* row = xh + (size_t)idx * DIM_I;
    const _Float16* w   = &sWw[half * DIM_I];

    float p = 0.f;
    #pragma unroll
    for (int j = 0; j < DIM_I; j += 8) {
        v8h hv = *(const v8h*)(row + j);
        v8h wf = *(const v8h*)(w + j);
        #pragma unroll
        for (int i = 0; i < 8; ++i) p += (float)hv[i] * (float)wf[i];
    }
    p += __shfl_xor(p, 1, 32);
    if (half == 0) {
        const float aval = p + bw[0];
        a_arr[e] = aval;
        atomicAdd(&denom[tgt[e]], aval);
    }
}

// ---------------------------------------------------------------------------
// Kernel 4: persistent-wave WMMA edge kernel (lean: no gate math).
//   Preload 16 register-resident B operands (Wf) once, then loop 5 tiles of
//   16 edges: A-fragments straight from global xh, 16 WMMAs, relu +
//   precomputed-gate scale, atomic segment-sum scatter into d_out.
// ---------------------------------------------------------------------------
__global__ __launch_bounds__(256)
void gat_edge_wmma(const _Float16* __restrict__ xh,
                   const int*   __restrict__ src,
                   const int*   __restrict__ tgt,
                   const float* __restrict__ Wf,
                   const float* __restrict__ bf,
                   const float* __restrict__ a_arr,
                   float* __restrict__ outAcc) {
    __shared__ __attribute__((aligned(16))) _Float16 sWf[DIM_O * KDIM]; // 16 KB
    __shared__ float sA[NUM_WAVES][16];
    __shared__ int   sTgt[NUM_WAVES][16];

    const int tid = threadIdx.x;
    for (int i = tid; i < DIM_O * KDIM; i += 256) sWf[i] = (_Float16)Wf[i];
    __syncthreads();

    const int wv   = tid >> 5;
    const int lane = tid & 31;
    const int mrow = lane & 15;           // A row / B column within 16-block
    const int hi   = lane >> 4;           // lane half selects K sub-range

    // ---- register-resident B operands: Bop[kb*4+nb], column n = nb*16+mrow,
    //      K = kb*32 + (hi?16:0) .. +15 contiguous halfs of Wf row n ---------
    v16h Bop[16];
    #pragma unroll
    for (int kb = 0; kb < 4; ++kb) {
        #pragma unroll
        for (int nb = 0; nb < 4; ++nb) {
            const _Float16* brow = &sWf[(nb * 16 + mrow) * KDIM + kb * 32 + hi * 16];
            v8h b0 = *(const v8h*)(brow);
            v8h b1 = *(const v8h*)(brow + 8);
            v16h B;
            #pragma unroll
            for (int i = 0; i < 8; ++i) { B[i] = b0[i]; B[i + 8] = b1[i]; }
            Bop[kb * 4 + nb] = B;
        }
    }
    float biasv[4];
    #pragma unroll
    for (int nb = 0; nb < 4; ++nb) biasv[nb] = bf[nb * 16 + mrow];

    const int waveId = blockIdx.x * NUM_WAVES + wv;

    #pragma unroll 1
    for (int itr = 0; itr < TILES_PER_WAVE; ++itr) {
        const int tile  = waveId + itr * TOT_WAVES;
        const int eBase = tile * 16;
        // lane handles row m = mrow of the tile
        const int iS = src[eBase + mrow];
        const int iT = tgt[eBase + mrow];
        const _Float16* rowS = xh + (size_t)iS * DIM_I;
        const _Float16* rowT = xh + (size_t)iT * DIM_I;

        if (lane < 16) {
            sA[wv][lane]   = a_arr[eBase + lane];   // gate a[m], m = lane
            sTgt[wv][lane] = iT;                    // tgt row for edge m = lane
        }

        v8f acc[4];
        #pragma unroll
        for (int nb = 0; nb < 4; ++nb)
            #pragma unroll
            for (int r = 0; r < 8; ++r) acc[nb][r] = biasv[nb];

        #pragma unroll
        for (int kb = 0; kb < 4; ++kb) {
            // A fragment: kb<2 -> src half (K 0..63), kb>=2 -> tgt half
            const _Float16* row = (kb < 2) ? rowS : rowT;
            const int o0 = (kb & 1) * 32 + (hi ? 16 : 0);
            const int o1 = (kb & 1) * 32 + (hi ? 24 : 8);
            v8h a0 = *(const v8h*)(row + o0);       // global b128, 16B aligned
            v8h a1 = *(const v8h*)(row + o1);
            v16h A;
            #pragma unroll
            for (int i = 0; i < 8; ++i) { A[i] = a0[i]; A[i + 8] = a1[i]; }

            #pragma unroll
            for (int nb = 0; nb < 4; ++nb)
                acc[nb] = __builtin_amdgcn_wmma_f32_16x16x32_f16(
                    false, A, false, Bop[kb * 4 + nb], (short)0, acc[nb],
                    false, false);
        }

        // relu, gate-scale, atomic segment-sum scatter (L2-resident target)
        #pragma unroll
        for (int nb = 0; nb < 4; ++nb) {
            const int n = nb * 16 + mrow;
            #pragma unroll
            for (int r = 0; r < 8; ++r) {
                const int m = r + hi * 8;     // D layout: lanes>=16 hold M=8..15
                float y = acc[nb][r];
                y = y > 0.f ? y : 0.f;
                atomicAdd(&outAcc[(size_t)sTgt[wv][m] * DIM_O + n],
                          y * sA[wv][m]);
            }
        }
    }
}

// ---------------------------------------------------------------------------
// Kernel 5: o = numer / (denom + eps)
// ---------------------------------------------------------------------------
__global__ void gat_final(float* __restrict__ out, const float* __restrict__ denom) {
    int i = blockIdx.x * blockDim.x + threadIdx.x;
    if (i < N_NODES * DIM_O) out[i] = out[i] / (denom[i >> 6] + GAT_EPS);
}

// ---------------------------------------------------------------------------
extern "C" void kernel_launch(void* const* d_in, const int* in_sizes, int n_in,
                              void* d_out, int out_size, void* d_ws, size_t ws_size,
                              hipStream_t stream) {
    (void)in_sizes; (void)n_in; (void)out_size; (void)ws_size;
    const float* x   = (const float*)d_in[0];
    const int*   src = (const int*)  d_in[2];
    const int*   tgt = (const int*)  d_in[3];
    const float* Wf  = (const float*)d_in[5];
    const float* bf  = (const float*)d_in[6];
    const float* Ww  = (const float*)d_in[7];
    const float* bw  = (const float*)d_in[8];
    float* out = (float*)d_out;

    _Float16* xh    = (_Float16*)d_ws;                               // 6.4 MB
    float*    a_arr = (float*)((char*)d_ws + XH_BYTES);              // 3.2 MB
    float*    denom = (float*)((char*)d_ws + XH_BYTES + A_BYTES);    // 0.2 MB

    const int tot = N_NODES * DIM_O;
    gat_init<<<(tot + 255) / 256, 256, 0, stream>>>(out, denom);
    gat_x2h <<<(N_NODES * DIM_I + 255) / 256, 256, 0, stream>>>(x, xh);
    gat_gate<<<(2 * N_EDGES) / 256, 256, 0, stream>>>(xh, src, tgt, Ww, bw,
                                                      a_arr, denom);
    gat_edge_wmma<<<GRID_BLKS, 256, 0, stream>>>(xh, src, tgt, Wf, bf,
                                                 a_arr, out);
    gat_final<<<(tot + 255) / 256, 256, 0, stream>>>(out, denom);
}
